// MultiHeadAttentionWithCombinedQKV_71219147702592
// MI455X (gfx1250) — matrix operand order repaired and checked by reading, not compiled, not run on hardware
//
#include <hip/hip_runtime.h>

typedef __attribute__((ext_vector_type(16))) _Float16 v16h;
typedef __attribute__((ext_vector_type(8)))  float    v8f;
typedef unsigned int u32;

#define NQ     4096
#define DMODEL 512
#define NHEAD  8
#define DH     64
#define ROWS   (2 * NQ)   // B*N = 8192

// ---------------------------------------------------------------------------
// CDNA5 async tile mover: global -> LDS, 16B/lane, tracked by ASYNCcnt
// ---------------------------------------------------------------------------
__device__ __forceinline__ u32 lds_addr32(const void* p) {
    return (u32)(uintptr_t)p;   // LDS aperture: addr[31:0] = LDS byte offset
}
__device__ __forceinline__ void async_b128(u32 lds, const void* gsrc) {
    asm volatile("global_load_async_to_lds_b128 %0, %1, off"
                 :: "v"(lds), "v"(gsrc) : "memory");
}
#define WAIT_ASYNC(n) asm volatile("s_wait_asynccnt " #n ::: "memory")

// ---------------------------------------------------------------------------
// Fragment loaders per CDNA5 ISA 7.12.2 (wave32, 16-bit A 16x32, B 32x16)
// A: lane(0-15)=row M, VGPR i: k = 2*(i&3) + 16*(i>>2) + 8*hi
// B: lane(0-15)=col N, VGPR i: k = 2*(i&3) + 16*(i>>2) + 8*hi
// (each vectorizes to 2x ds_load_b128)
// ---------------------------------------------------------------------------
__device__ __forceinline__ v16h frag_a(const _Float16* p, int stride, int m, int hi) {
    union { v16h v; u32 u[8]; } f;
#pragma unroll
    for (int i = 0; i < 8; ++i) {
        int k = 2 * (i & 3) + ((i >> 2) << 4) + (hi << 3);
        f.u[i] = *(const u32*)(p + m * stride + k);
    }
    return f.v;
}
__device__ __forceinline__ v16h frag_b(const _Float16* p, int stride, int n, int hi) {
    union { v16h v; u32 u[8]; } f;
#pragma unroll
    for (int i = 0; i < 8; ++i) {
        int k = 2 * (i & 3) + ((i >> 2) << 4) + (hi << 3);
        f.u[i] = *(const u32*)(p + n * stride + k);
    }
    return f.v;
}

#define WMMA_F16(A, B, C) \
    __builtin_amdgcn_wmma_f32_16x16x32_f16(false, (A), false, (B), (short)0, (C), false, false)

// ---------------------------------------------------------------------------
// fp32 -> f16, 8 elements per thread
// ---------------------------------------------------------------------------
__global__ void cvt_f32_f16_kernel(const float* __restrict__ src,
                                   _Float16* __restrict__ dst, int n8) {
    int i = blockIdx.x * blockDim.x + threadIdx.x;
    if (i < n8) {
        const float4* s = (const float4*)src;
        float4 a = s[2 * i], b = s[2 * i + 1];
        union { uint4 q; _Float16 h[8]; } o;
        o.h[0] = (_Float16)a.x; o.h[1] = (_Float16)a.y;
        o.h[2] = (_Float16)a.z; o.h[3] = (_Float16)a.w;
        o.h[4] = (_Float16)b.x; o.h[5] = (_Float16)b.y;
        o.h[6] = (_Float16)b.z; o.h[7] = (_Float16)b.w;
        ((uint4*)dst)[i] = o.q;
    }
}

// ---------------------------------------------------------------------------
// Async double-buffered 128x64 GEMM mainloop (K=512, k-step 32).
// Each of 4 waves owns 32 rows: 2 A-frags x 4 B-frags = 8 WMMA / k-step.
// acc[h*4+t] covers rows (wv*32 + h*16), cols (t*16).
// ---------------------------------------------------------------------------
__device__ __forceinline__ void gemm128x64_mainloop(
    const _Float16* __restrict__ aSrc, const _Float16* __restrict__ bSrc,
    _Float16* At, _Float16* Bt, int tid, v8f acc[8]) {
    const int lane = tid & 31, wv = tid >> 5;
    const int hi = lane >> 4, ln = lane & 15;

    auto issue = [&](int buf, int k0) {
        _Float16* Ab = At + buf * 4096;   // 128 x 32 halfs
        _Float16* Bb = Bt + buf * 2048;   //  64 x 32 halfs
#pragma unroll
        for (int i = 0; i < 4; ++i) {     // A: 512 uint4
            int u = tid + (i << 7);
            int r = u >> 2, c = u & 3;
            async_b128(lds_addr32(Ab) + u * 16, aSrc + (size_t)r * DMODEL + k0 + c * 8);
        }
#pragma unroll
        for (int i = 0; i < 2; ++i) {     // B: 256 uint4
            int u = tid + (i << 7);
            int r = u >> 2, c = u & 3;
            async_b128(lds_addr32(Bb) + u * 16, bSrc + (size_t)r * DMODEL + k0 + c * 8);
        }
    };

    issue(0, 0);
    for (int it = 0; it < 16; ++it) {
        int buf = it & 1;
        if (it < 15) { issue(buf ^ 1, (it + 1) * 32); WAIT_ASYNC(0x6); }
        else         { WAIT_ASYNC(0x0); }
        __syncthreads();
        const _Float16* Ab = At + buf * 4096 + wv * 32 * 32;
        const _Float16* Bb = Bt + buf * 2048;
        v16h af0 = frag_a(Ab, 32, ln, hi);
        v16h af1 = frag_a(Ab + 16 * 32, 32, ln, hi);
#pragma unroll
        for (int t = 0; t < 4; ++t) {
            v16h bf = frag_b(Bb + t * 16 * 32, 32, ln, hi);
            acc[t]     = WMMA_F16(af0, bf, acc[t]);
            acc[4 + t] = WMMA_F16(af1, bf, acc[4 + t]);
        }
        __syncthreads();
    }
}

// ---------------------------------------------------------------------------
// QKV projection -> Q (pre-scaled 1/sqrt(Dh)) [bh][n][dh], K [bh][n][dh],
// V pre-TRANSPOSED [bh][dh][n]. grid = (ROWS/128, 1536/64), block 128.
// ---------------------------------------------------------------------------
__global__ __launch_bounds__(128) void qkv_gemm_kernel(
    const _Float16* __restrict__ xh, const _Float16* __restrict__ wh,
    _Float16* __restrict__ qh, _Float16* __restrict__ kh,
    _Float16* __restrict__ vh) {
    __shared__ __align__(16) _Float16 At[2 * 4096];
    __shared__ __align__(16) _Float16 Bt[2 * 2048];

    const int tid = threadIdx.x;
    const int lane = tid & 31, wv = tid >> 5;
    const int hi = lane >> 4, ln = lane & 15;
    const int bM = blockIdx.x, bN = blockIdx.y;

    v8f zero = {};
    v8f acc[8];
#pragma unroll
    for (int t = 0; t < 8; ++t) acc[t] = zero;

    gemm128x64_mainloop(xh + (size_t)bM * 128 * DMODEL,
                        wh + (size_t)bN * 64 * DMODEL, At, Bt, tid, acc);

    const int s = bN >> 3;          // 0=Q 1=K 2=V (64-wide N block = one (s,h))
    const int h = bN & 7;
    const float scl = (s == 0) ? 0.125f : 1.0f;
#pragma unroll
    for (int half = 0; half < 2; ++half) {
#pragma unroll
        for (int t = 0; t < 4; ++t) {
#pragma unroll
            for (int r = 0; r < 8; ++r) {
                int row  = bM * 128 + wv * 32 + half * 16 + r + 8 * hi;
                int b    = row >> 12;
                int nseq = row & (NQ - 1);
                int dh   = t * 16 + ln;
                int bh   = b * NHEAD + h;
                _Float16 val = (_Float16)(acc[half * 4 + t][r] * scl);
                if (s == 0)      qh[(((size_t)bh * NQ + nseq) << 6) + dh] = val;
                else if (s == 1) kh[(((size_t)bh * NQ + nseq) << 6) + dh] = val;
                else             vh[(((size_t)bh << 6) + dh) * NQ + nseq] = val;
            }
        }
    }
}

// ---------------------------------------------------------------------------
// Causal flash attention. grid = (NQ/64, B*H), block 128 (4 waves).
// kv blocks of 64: 16 WMMA per block, ONE softmax pass per 64 kv.
// ---------------------------------------------------------------------------
__global__ __launch_bounds__(128) void attn_kernel(
    const _Float16* __restrict__ qh, const _Float16* __restrict__ kh,
    const _Float16* __restrict__ vh, _Float16* __restrict__ ctxh) {
    __shared__ __align__(16) _Float16 Kt[2 * 64 * 64];   // [kv][dh]
    __shared__ __align__(16) _Float16 Vt[2 * 64 * 64];   // [dh][kv]
    __shared__ __align__(16) _Float16 Pl[4 * 16 * 64];   // per-wave P tile

    const int tid = threadIdx.x;
    const int lane = tid & 31, wv = tid >> 5;
    const int hi = lane >> 4, ln = lane & 15;
    const int qblk = blockIdx.x, bh = blockIdx.y;
    const int qbase = qblk * 64 + wv * 16;

    // Q fragments (constant over kv loop)
    const _Float16* qptr = qh + ((size_t)(bh * NQ + qbase) << 6);
    union { v16h v; u32 u[8]; } aq0, aq1;
#pragma unroll
    for (int i = 0; i < 8; ++i) {
        int k = 2 * (i & 3) + ((i >> 2) << 4) + (hi << 3);
        aq0.u[i] = *(const u32*)(qptr + ln * DH + k);
        aq1.u[i] = *(const u32*)(qptr + ln * DH + k + 32);
    }

    const _Float16* kbase = kh + ((size_t)bh * NQ << 6);   // [n][dh]
    const _Float16* vbase = vh + ((size_t)bh << 6) * NQ;   // [dh][n]
    auto issue_tiles = [&](int buf, int kv) {
        _Float16* Kb = Kt + buf * 4096;
        _Float16* Vb = Vt + buf * 4096;
        const _Float16* ks = kbase + ((size_t)kv << 6);    // contiguous 8KB
#pragma unroll
        for (int i = 0; i < 4; ++i) {
            int u = tid + (i << 7);                        // uint4 slot 0..511
            async_b128(lds_addr32(Kb) + u * 16, ks + u * 8);
            int r = u >> 3, c = u & 7;                     // V: 64 dh rows x 8
            async_b128(lds_addr32(Vb) + u * 16, vbase + (size_t)r * NQ + kv + c * 8);
        }
    };

    v8f zero = {};
    v8f accv[4];
#pragma unroll
    for (int t = 0; t < 4; ++t) accv[t] = zero;
    float mrow[8], lrow[8];
#pragma unroll
    for (int r = 0; r < 8; ++r) { mrow[r] = -1e30f; lrow[r] = 0.0f; }

    const int nIt = qblk + 1;            // causal: 64-wide kv blocks
    issue_tiles(0, 0);
    for (int it = 0; it < nIt; ++it) {
        const int buf = it & 1;
        const int kv = it * 64;
        if (it + 1 < nIt) { issue_tiles(buf ^ 1, kv + 64); WAIT_ASYNC(0x8); }
        else              { WAIT_ASYNC(0x0); }
        __syncthreads();
        const _Float16* Kb = Kt + buf * 4096;
        const _Float16* Vb = Vt + buf * 4096;

        // S(16x64) = Q @ K^T : four 16-wide N tiles, contraction 64
        v8f sc[4];
#pragma unroll
        for (int t = 0; t < 4; ++t) {
            sc[t] = zero;
            v16h b0 = frag_b(Kb + t * 16 * DH, DH, ln, hi);
            v16h b1 = frag_b(Kb + t * 16 * DH + 32, DH, ln, hi);
            sc[t] = WMMA_F16(aq0.v, b0, sc[t]);
            sc[t] = WMMA_F16(aq1.v, b1, sc[t]);
        }

        // causal mask in place
#pragma unroll
        for (int t = 0; t < 4; ++t) {
            const int n = kv + t * 16 + ln;
#pragma unroll
            for (int r = 0; r < 8; ++r) {
                int me = qbase + r + 8 * hi;
                if (n > me) sc[t][r] = -1e30f;
            }
        }

        // one online-softmax pass per 64 kv (row = vgpr r + 8*hi)
#pragma unroll
        for (int r = 0; r < 8; ++r) {
            float mx = fmaxf(fmaxf(sc[0][r], sc[1][r]), fmaxf(sc[2][r], sc[3][r]));
            mx = fmaxf(mx, __shfl_xor(mx, 1, 32));
            mx = fmaxf(mx, __shfl_xor(mx, 2, 32));
            mx = fmaxf(mx, __shfl_xor(mx, 4, 32));
            mx = fmaxf(mx, __shfl_xor(mx, 8, 32));
            float mnew = fmaxf(mrow[r], mx);
            float corr = __expf(mrow[r] - mnew);
            float p0 = __expf(sc[0][r] - mnew);
            float p1 = __expf(sc[1][r] - mnew);
            float p2 = __expf(sc[2][r] - mnew);
            float p3 = __expf(sc[3][r] - mnew);
            float rs = (p0 + p1) + (p2 + p3);
            rs += __shfl_xor(rs, 1, 32);
            rs += __shfl_xor(rs, 2, 32);
            rs += __shfl_xor(rs, 4, 32);
            rs += __shfl_xor(rs, 8, 32);
            lrow[r] = lrow[r] * corr + rs;
            mrow[r] = mnew;
#pragma unroll
            for (int t = 0; t < 4; ++t) accv[t][r] *= corr;
            _Float16* prow = Pl + wv * 1024 + (r + 8 * hi) * 64 + ln;
            prow[0]  = (_Float16)p0;
            prow[16] = (_Float16)p1;
            prow[32] = (_Float16)p2;
            prow[48] = (_Float16)p3;
        }

        // ctx += P(16x64) @ V(64x64)
        v16h pf0 = frag_a(Pl + wv * 1024, 64, ln, hi);
        v16h pf1 = frag_a(Pl + wv * 1024 + 32, 64, ln, hi);
#pragma unroll
        for (int t = 0; t < 4; ++t) {
            v16h bv0 = frag_b(Vb + t * 16 * 64, 64, ln, hi);
            v16h bv1 = frag_b(Vb + t * 16 * 64 + 32, 64, ln, hi);
            accv[t] = WMMA_F16(pf0, bv0, accv[t]);
            accv[t] = WMMA_F16(pf1, bv1, accv[t]);
        }
        __syncthreads();
    }

    // finalize: ctx/l -> ctxh[b][nseq][h*64+dh]
    const int b = bh >> 3, h = bh & 7;
#pragma unroll
    for (int r = 0; r < 8; ++r) {
        float inv = 1.0f / lrow[r];
        int nseq = qbase + r + 8 * hi;
#pragma unroll
        for (int t = 0; t < 4; ++t) {
            int dh = t * 16 + ln;
            ctxh[((size_t)(b * NQ + nseq) << 9) + h * 64 + dh] =
                (_Float16)(accv[t][r] * inv);
        }
    }
}

// ---------------------------------------------------------------------------
// Output projection: out = ctx @ Wproj^T + bias (fp32). grid=(64,8), block 128.
// ---------------------------------------------------------------------------
__global__ __launch_bounds__(128) void proj_gemm_kernel(
    const _Float16* __restrict__ ch, const _Float16* __restrict__ wh,
    const float* __restrict__ bias, float* __restrict__ out) {
    __shared__ __align__(16) _Float16 At[2 * 4096];
    __shared__ __align__(16) _Float16 Bt[2 * 2048];

    const int tid = threadIdx.x;
    const int lane = tid & 31, wv = tid >> 5;
    const int hi = lane >> 4, ln = lane & 15;
    const int bM = blockIdx.x, bN = blockIdx.y;

    v8f zero = {};
    v8f acc[8];
#pragma unroll
    for (int t = 0; t < 8; ++t) acc[t] = zero;

    gemm128x64_mainloop(ch + (size_t)bM * 128 * DMODEL,
                        wh + (size_t)bN * 64 * DMODEL, At, Bt, tid, acc);

#pragma unroll
    for (int half = 0; half < 2; ++half) {
#pragma unroll
        for (int t = 0; t < 4; ++t) {
#pragma unroll
            for (int r = 0; r < 8; ++r) {
                int row = bM * 128 + wv * 32 + half * 16 + r + 8 * hi;
                int j   = bN * 64 + t * 16 + ln;
                out[(size_t)row * DMODEL + j] = acc[half * 4 + t][r] + bias[j];
            }
        }
    }
}

// ---------------------------------------------------------------------------
extern "C" void kernel_launch(void* const* d_in, const int* in_sizes, int n_in,
                              void* d_out, int out_size, void* d_ws, size_t ws_size,
                              hipStream_t stream) {
    const float* x      = (const float*)d_in[0];
    const float* w_qkv  = (const float*)d_in[1];
    const float* w_proj = (const float*)d_in[2];
    const float* b_proj = (const float*)d_in[3];
    float* out = (float*)d_out;

    const size_t N_X    = (size_t)ROWS * DMODEL;        // 4,194,304
    const size_t N_WQKV = (size_t)3 * DMODEL * DMODEL;  //   786,432
    const size_t N_WPRJ = (size_t)DMODEL * DMODEL;      //   262,144

    _Float16* xh     = (_Float16*)d_ws;
    _Float16* wqkvh  = xh + N_X;
    _Float16* wprojh = wqkvh + N_WQKV;
    _Float16* qh     = wprojh + N_WPRJ;
    _Float16* kh     = qh + N_X;
    _Float16* vh     = kh + N_X;
    _Float16* ctxh   = vh + N_X;

    cvt_f32_f16_kernel<<<(int)(N_X / 8 + 255) / 256, 256, 0, stream>>>(x, xh, (int)(N_X / 8));
    cvt_f32_f16_kernel<<<(int)(N_WQKV / 8 + 255) / 256, 256, 0, stream>>>(w_qkv, wqkvh, (int)(N_WQKV / 8));
    cvt_f32_f16_kernel<<<(int)(N_WPRJ / 8 + 255) / 256, 256, 0, stream>>>(w_proj, wprojh, (int)(N_WPRJ / 8));

    qkv_gemm_kernel<<<dim3(ROWS / 128, (3 * DMODEL) / 64), 128, 0, stream>>>(
        xh, wqkvh, qh, kh, vh);

    attn_kernel<<<dim3(NQ / 64, 2 * NHEAD), 128, 0, stream>>>(qh, kh, vh, ctxh);

    proj_gemm_kernel<<<dim3(ROWS / 128, DMODEL / 64), 128, 0, stream>>>(
        ctxh, wprojh, b_proj, out);
}
